// MultiHeadAttention_23785528885514
// MI455X (gfx1250) — matrix operand order repaired and checked by reading, not compiled, not run on hardware
//
#include <hip/hip_runtime.h>
#include <cstdint>
#include <cstddef>

// ---------------------------------------------------------------------------
// MHA forward for MI455X (gfx1250, wave32, WMMA).
// - All matmuls on v_wmma_f32_16x16x32_bf16 (f32 accumulate).
// - GEMMs: double-buffered LDS staging via GLOBAL_LOAD_ASYNC_TO_LDS_B128
//   (ASYNCcnt pipeline), K-step 64 (8 WMMAs per barrier), conflict-free
//   padded LDS, ds_load_b128 fragments.
// - Attention: flash-style online softmax (no SxS materialization), V
//   fragments prefetched under the expf shadow.
// ---------------------------------------------------------------------------

typedef __attribute__((ext_vector_type(16))) __bf16 v16bf;
typedef __attribute__((ext_vector_type(8)))  float  v8f;

union Frag { v16bf v; uint4 u[2]; };

__device__ __forceinline__ v8f zero8() {
    v8f z;
#pragma unroll
    for (int i = 0; i < 8; ++i) z[i] = 0.0f;
    return z;
}

// Low 32 bits of a flat shared pointer == LDS byte offset.
__device__ __forceinline__ uint32_t lds_off(const void* p) {
    return (uint32_t)(uintptr_t)p;
}

// Async copy of 16B/lane from global (saddr + u32 voffset) into LDS.
#define ASYNC_B128(LDSOFF, GOFF, SBASE)                                      \
    asm volatile("global_load_async_to_lds_b128 %0, %1, %2"                  \
                 :: "v"(LDSOFF), "v"(GOFF), "s"(SBASE) : "memory")

#define WAIT_ASYNC0() asm volatile("s_wait_asynccnt 0x0" ::: "memory")

static constexpr int Bc   = 2;
static constexpr int Sq   = 2048;
static constexpr int Dm   = 1024;
static constexpr int Hh   = 16;
static constexpr int Dh   = 64;
static constexpr int Mrow = Bc * Sq;           // 4096

// ------------------------------ fp32 -> bf16 -------------------------------
__global__ __launch_bounds__(256) void cvt_bf16_kernel(
    const float* __restrict__ src, __bf16* __restrict__ dst, int n)
{
    int i = blockIdx.x * 256 + threadIdx.x;
    if (i < n) dst[i] = (__bf16)src[i];
}

// -------------------- transpose + convert weight to Wt[n][k] ---------------
__global__ __launch_bounds__(256) void transpose_cvt_kernel(
    const float* __restrict__ w, __bf16* __restrict__ wt)
{
    __shared__ float tile[32][33];
    const int bx = blockIdx.x * 32;   // n base
    const int by = blockIdx.y * 32;   // k base
    const int tx = threadIdx.x, ty = threadIdx.y;   // 32 x 8
#pragma unroll
    for (int i = 0; i < 32; i += 8)
        tile[ty + i][tx] = w[(by + ty + i) * Dm + (bx + tx)];
    __syncthreads();
#pragma unroll
    for (int i = 0; i < 32; i += 8)
        wt[(bx + ty + i) * Dm + (by + tx)] = (__bf16)tile[tx][ty + i];
}

// ------------------------------- WMMA GEMM ---------------------------------
// C[m][n] = sum_k A[m][k] * Wt[n][k] + bias[n]
// MODE 0: -> Q  bf16 [B,H,S,Dh]   MODE 1: -> K bf16 [B,H,S,Dh]
// MODE 2: -> Vt bf16 [B,H,Dh,S]   MODE 3: -> out f32 [M,N]
template <int MODE>
__global__ __launch_bounds__(256) void gemm_bf16_kernel(
    const __bf16* __restrict__ A, const __bf16* __restrict__ WT,
    const float* __restrict__ bias, void* __restrict__ outp)
{
    // K-step 64; rows padded to 72 elems (36 dwords) -> conflict-free b128.
    __shared__ __align__(16) __bf16 Asm[2][128][72];
    __shared__ __align__(16) __bf16 Bsm[2][64][72];

    const int tid  = threadIdx.x;
    const int w    = tid >> 5, lane = tid & 31;
    const int l16  = lane & 15, half = lane >> 4, kb = half * 8;
    const int wm   = w >> 1, wn = w & 1;
    const int m0b  = blockIdx.x * 128;
    const int n0b  = blockIdx.y * 64;

    // Staging: A tile 128x64 = 1024 16B-chunks (4/thread),
    //          B tile  64x64 =  512 16B-chunks (2/thread)
    const int arow = tid >> 1;                 // A row (4 chunks, same row)
    const int akc  = (tid & 1) * 32;           // first of 4x8-elem chunks
    const int brow = tid >> 2;                 // B row (2 chunks, same row)
    const int bkc  = (tid & 3) * 16;           // first of 2x8-elem chunks

    const uint64_t Abase = (uint64_t)(uintptr_t)A;
    const uint64_t Wbase = (uint64_t)(uintptr_t)WT;

    auto stage = [&](int buf, int k0) {
#pragma unroll
        for (int j = 0; j < 4; ++j) {
            const int kc = akc + j * 8;
            uint32_t ga = (uint32_t)(((m0b + arow) * Dm + k0 + kc) * 2);
            ASYNC_B128(lds_off(&Asm[buf][arow][kc]), ga, Abase);
        }
#pragma unroll
        for (int j = 0; j < 2; ++j) {
            const int kc = bkc + j * 8;
            uint32_t ga = (uint32_t)(((n0b + brow) * Dm + k0 + kc) * 2);
            ASYNC_B128(lds_off(&Bsm[buf][brow][kc]), ga, Wbase);
        }
    };

    v8f acc[2][2];
#pragma unroll
    for (int im = 0; im < 2; ++im)
#pragma unroll
        for (int in = 0; in < 2; ++in) acc[im][in] = zero8();

    stage(0, 0);

    for (int k0 = 0, it = 0; k0 < Dm; k0 += 64, ++it) {
        const int buf = it & 1;
        WAIT_ASYNC0();          // my slab for `buf` has landed in LDS
        __syncthreads();        // everyone's landed; everyone done reading buf^1
        if (k0 + 64 < Dm) stage(buf ^ 1, k0 + 64);   // prefetch next slab

        Frag a[2][2], b[2][2];
#pragma unroll
        for (int ks = 0; ks < 2; ++ks) {
#pragma unroll
            for (int im = 0; im < 2; ++im) {
                const __bf16* ap = &Asm[buf][wm * 32 + im * 16 + l16][ks * 32];
                a[ks][im].u[0] = *(const uint4*)(ap + half * 16);
                a[ks][im].u[1] = *(const uint4*)(ap + half * 16 + 8);
            }
#pragma unroll
            for (int in = 0; in < 2; ++in) {
                const __bf16* bp =
                    &Bsm[buf][wn * 32 + in * 16 + l16][ks * 32 + half * 16];
                b[ks][in].u[0] = *(const uint4*)(bp);
                b[ks][in].u[1] = *(const uint4*)(bp + 8);
            }
        }
#pragma unroll
        for (int ks = 0; ks < 2; ++ks)
#pragma unroll
            for (int im = 0; im < 2; ++im)
#pragma unroll
                for (int in = 0; in < 2; ++in)
                    acc[im][in] = __builtin_amdgcn_wmma_f32_16x16x32_bf16(
                        false, a[ks][im].v, false, b[ks][in].v, (short)0,
                        acc[im][in], false, false);
    }

#pragma unroll
    for (int im = 0; im < 2; ++im) {
#pragma unroll
        for (int in = 0; in < 2; ++in) {
            const int col = n0b + wn * 32 + in * 16 + l16;
            const float bv = bias[col];
#pragma unroll
            for (int v = 0; v < 8; ++v) {
                const int row = m0b + wm * 32 + im * 16 + v + half * 8;
                const float val = acc[im][in][v] + bv;
                if constexpr (MODE == 3) {
                    ((float*)outp)[(size_t)row * Dm + col] = val;
                } else {
                    const int bi = row >> 11, s = row & (Sq - 1);
                    const int h = col >> 6,  d = col & (Dh - 1);
                    __bf16* o = (__bf16*)outp;
                    if constexpr (MODE == 2)  // V transposed [B,H,Dh,S]
                        o[(((size_t)(bi * Hh + h) * Dh + d) * Sq) + s] = (__bf16)val;
                    else                      // Q/K [B,H,S,Dh]
                        o[(((size_t)(bi * Hh + h) * Sq + s) * Dh) + d] = (__bf16)val;
                }
            }
        }
    }
}

// ----------------------------- flash attention -----------------------------
// 8 waves/block, 16 query rows per wave, online softmax over 64-key blocks.
__global__ __launch_bounds__(256) void flash_attn_kernel(
    const __bf16* __restrict__ Q, const __bf16* __restrict__ K,
    const __bf16* __restrict__ Vt, __bf16* __restrict__ ctx)
{
    const int w    = threadIdx.x >> 5;
    const int lane = threadIdx.x & 31;
    const int l16  = lane & 15;
    const int half = lane >> 4;
    const int kbse = half * 8;

    const int bi = blockIdx.z, h = blockIdx.y;
    const int q0 = blockIdx.x * 128 + w * 16;

    const __bf16* Qb = Q  + (size_t)(bi * Hh + h) * Sq * Dh;
    const __bf16* Kb = K  + (size_t)(bi * Hh + h) * Sq * Dh;
    const __bf16* Vb = Vt + (size_t)(bi * Hh + h) * Dh * Sq;

    __shared__ __align__(16) float  sc[8][16][68];     // scores, padded
    __shared__ __align__(16) __bf16 pb[8][16][72];     // probs bf16, padded
    __shared__ float mrow[8][16], lrow[8][16], frow[8][16];

    // Q fragments for this wave's 16 rows (two K=32 slabs over Dh=64)
    Frag aq[2];
    {
        const __bf16* qp = Qb + (size_t)(q0 + l16) * Dh;
#pragma unroll
        for (int t = 0; t < 2; ++t) {
            aq[t].u[0] = *(const uint4*)(qp + t * 32 + half * 16);
            aq[t].u[1] = *(const uint4*)(qp + t * 32 + half * 16 + 8);
        }
    }

    v8f oacc[4];
#pragma unroll
    for (int t = 0; t < 4; ++t) oacc[t] = zero8();

    if (lane < 16) { mrow[w][lane] = -1.0e30f; lrow[w][lane] = 0.0f; }

    for (int kblk = 0; kblk < Sq; kblk += 64) {
        // ---- scores: S = Q K^T * (1/sqrt(Dh)) for 16q x 64keys ----
#pragma unroll
        for (int n = 0; n < 4; ++n) {
            Frag bk0, bk1;
            const __bf16* kp = Kb + (size_t)(kblk + 16 * n + l16) * Dh + half * 16;
            bk0.u[0] = *(const uint4*)(kp);
            bk0.u[1] = *(const uint4*)(kp + 8);
            bk1.u[0] = *(const uint4*)(kp + 32);
            bk1.u[1] = *(const uint4*)(kp + 40);
            v8f c = zero8();
            c = __builtin_amdgcn_wmma_f32_16x16x32_bf16(
                    false, aq[0].v, false, bk0.v, (short)0, c, false, false);
            c = __builtin_amdgcn_wmma_f32_16x16x32_bf16(
                    false, aq[1].v, false, bk1.v, (short)0, c, false, false);
#pragma unroll
            for (int v = 0; v < 8; ++v)
                sc[w][v + half * 8][16 * n + l16] = c[v] * 0.125f;
        }

        // ---- prefetch V fragments (k0=0) under the softmax shadow ----
        Frag bvf[2][4];
#pragma unroll
        for (int n = 0; n < 4; ++n) {
            const __bf16* vp = Vb + (size_t)(16 * n + l16) * Sq + kblk + half * 16;
            bvf[0][n].u[0] = *(const uint4*)(vp);
            bvf[0][n].u[1] = *(const uint4*)(vp + 8);
        }

        // ---- online softmax (lanes 0..15 own one row each) ----
        if (lane < 16) {
            const float mo = mrow[w][lane], lo = lrow[w][lane];
            float mx = mo;
#pragma unroll 8
            for (int j = 0; j < 64; ++j) mx = fmaxf(mx, sc[w][lane][j]);
            float ladd = 0.0f;
#pragma unroll 8
            for (int j = 0; j < 64; ++j) {
                const float p = __expf(sc[w][lane][j] - mx);
                pb[w][lane][j] = (__bf16)p;
                ladd += p;
            }
            const float fac = __expf(mo - mx);
            mrow[w][lane] = mx;
            lrow[w][lane] = lo * fac + ladd;
            frow[w][lane] = fac;
        }

        // ---- rescale running output ----
        float fv[8];
#pragma unroll
        for (int v = 0; v < 8; ++v) fv[v] = frow[w][v + half * 8];
#pragma unroll
        for (int t = 0; t < 4; ++t)
#pragma unroll
            for (int v = 0; v < 8; ++v) oacc[t][v] *= fv[v];

        // ---- O += P V  (two K=32 key slabs, 4 d-tiles) ----
#pragma unroll
        for (int k0 = 0; k0 < 64; k0 += 32) {
            const int cb = k0 >> 5;
            if (k0 == 0) {   // prefetch k0=32 set under the first WMMAs
#pragma unroll
                for (int n = 0; n < 4; ++n) {
                    const __bf16* vp =
                        Vb + (size_t)(16 * n + l16) * Sq + kblk + 32 + half * 16;
                    bvf[1][n].u[0] = *(const uint4*)(vp);
                    bvf[1][n].u[1] = *(const uint4*)(vp + 8);
                }
            }
            Frag ap;
            const __bf16* pp = &pb[w][l16][k0 + half * 16];
            ap.u[0] = *(const uint4*)(pp);
            ap.u[1] = *(const uint4*)(pp + 8);
#pragma unroll
            for (int n = 0; n < 4; ++n)
                oacc[n] = __builtin_amdgcn_wmma_f32_16x16x32_bf16(
                    false, ap.v, false, bvf[cb][n].v, (short)0, oacc[n],
                    false, false);
        }
    }

    // ---- normalize and store context in [B,S,H*Dh] (bf16) ----
    float linv[8];
#pragma unroll
    for (int v = 0; v < 8; ++v) linv[v] = 1.0f / lrow[w][v + half * 8];
#pragma unroll
    for (int t = 0; t < 4; ++t) {
#pragma unroll
        for (int v = 0; v < 8; ++v) {
            const int row = q0 + v + half * 8;
            const int d   = 16 * t + l16;
            const float val = oacc[t][v] * linv[v];
            ctx[((size_t)bi * Sq + row) * Dm + h * Dh + d] = (__bf16)val;
        }
    }
}

// ------------------------------- launcher ----------------------------------
extern "C" void kernel_launch(void* const* d_in, const int* in_sizes, int n_in,
                              void* d_out, int out_size, void* d_ws, size_t ws_size,
                              hipStream_t stream)
{
    const float* x  = (const float*)d_in[0];
    const float* wq = (const float*)d_in[1];
    const float* bq = (const float*)d_in[2];
    const float* wk = (const float*)d_in[3];
    const float* bk = (const float*)d_in[4];
    const float* wv = (const float*)d_in[5];
    const float* bv = (const float*)d_in[6];
    const float* wo = (const float*)d_in[7];
    const float* bo = (const float*)d_in[8];

    // Workspace layout (all bf16 elements): 24M elems = 48 MB
    __bf16* base = (__bf16*)d_ws;
    __bf16* xb  = base;                               // [4096,1024]
    __bf16* wqT = base + (size_t)4  * 1024 * 1024;    // [1024,1024] transposed
    __bf16* wkT = base + (size_t)5  * 1024 * 1024;
    __bf16* wvT = base + (size_t)6  * 1024 * 1024;
    __bf16* woT = base + (size_t)7  * 1024 * 1024;
    __bf16* Qd  = base + (size_t)8  * 1024 * 1024;    // [B,H,S,Dh]
    __bf16* Kd  = base + (size_t)12 * 1024 * 1024;    // [B,H,S,Dh]
    __bf16* Vtd = base + (size_t)16 * 1024 * 1024;    // [B,H,Dh,S]
    __bf16* ctx = base + (size_t)20 * 1024 * 1024;    // [B,S,D]

    const int nx = Mrow * Dm;                         // 4,194,304
    cvt_bf16_kernel<<<nx / 256, 256, 0, stream>>>(x, xb, nx);

    dim3 tb(32, 8), tg(32, 32);
    transpose_cvt_kernel<<<tg, tb, 0, stream>>>(wq, wqT);
    transpose_cvt_kernel<<<tg, tb, 0, stream>>>(wk, wkT);
    transpose_cvt_kernel<<<tg, tb, 0, stream>>>(wv, wvT);
    transpose_cvt_kernel<<<tg, tb, 0, stream>>>(wo, woT);

    dim3 gg(Mrow / 128, Dm / 64);   // 32 x 16
    gemm_bf16_kernel<0><<<gg, 256, 0, stream>>>(xb, wqT, bq, (void*)Qd);
    gemm_bf16_kernel<1><<<gg, 256, 0, stream>>>(xb, wkT, bk, (void*)Kd);
    gemm_bf16_kernel<2><<<gg, 256, 0, stream>>>(xb, wvT, bv, (void*)Vtd);

    flash_attn_kernel<<<dim3(Sq / 128, Hh, Bc), 256, 0, stream>>>(Qd, Kd, Vtd, ctx);

    gemm_bf16_kernel<3><<<gg, 256, 0, stream>>>(ctx, woT, bo, d_out);
}
